// ConceptNet_48206712930764
// MI455X (gfx1250) — matrix operand order repaired and checked by reading, not compiled, not run on hardware
//
#include <hip/hip_runtime.h>
#include <hip/hip_bf16.h>

// ---------------------------------------------------------------------------
// ConceptNet forward for MI455X (gfx1250, wave32, WMMA + TDM).
//  * Reassociate y_pred = (X@C) @ Ginv @ (C^T@W_h) + b  -> never build proj.
//  * GEMMs: TDM (tensor_load_to_lds) DMA fp32 tiles global->LDS, double
//    buffered on TENSORcnt; bf16 conversion LDS->VGPR; v_wmma_f32_16x16x32_bf16
//    with f32 accumulate. Problem is HBM-bound (~0.5GB @ 23.3TB/s), so the
//    DMA path + bf16 matrix density is the right operating point.
//  * bank read exactly once: R = bank@concept fused with row sum-of-squares.
//  * top-k on d2 (monotone in sqrt), deterministic LDS merge selection.
// ---------------------------------------------------------------------------

typedef __attribute__((ext_vector_type(16))) __bf16    v16bf;
typedef __attribute__((ext_vector_type(8)))  float     v8f;
typedef __attribute__((ext_vector_type(4)))  unsigned  v4u;
typedef __attribute__((ext_vector_type(8)))  int       v8i;
typedef __attribute__((ext_vector_type(4)))  int       v4i;

#define BSn   8192
#define Dn    1024
#define NCn   64
#define Nbank 100000
#define Ccls  1000
#define KSEL  50
#define FINF  3.402823466e38f

#if __has_builtin(__builtin_amdgcn_tensor_load_to_lds)
#define HAVE_TDM 1
#else
#define HAVE_TDM 0
#endif

__device__ __forceinline__ unsigned short f2bf(float f) {
  union { float f; unsigned u; } x; x.f = f;
  unsigned u = x.u;
  unsigned r = u + 0x7FFFu + ((u >> 16) & 1u);   // round-to-nearest-even
  return (unsigned short)(r >> 16);
}

#if HAVE_TDM
// LDS byte offset (workgroup-relative) of a __shared__ object.
__device__ __forceinline__ unsigned lds_addr_of(const void* p) {
  return (unsigned)(unsigned long long)(const __attribute__((address_space(3))) char*)p;
}

// 2-D TDM load: fp32 tile (tile0 x tile1) from row-major tensor into LDS,
// contiguous row-major in LDS. dims are REMAINING extents from the tile
// origin (TDM zero-fills out-of-bounds reads -> no edge guards needed).
// D# packing per CDNA5 ISA (Group0 128b, Group1 256b).
__device__ __forceinline__ void tdm_load_2d(unsigned ldsAddr, const void* gptr,
                                            unsigned dim0, unsigned dim1,
                                            unsigned stride0,
                                            unsigned tile0, unsigned tile1)
{
  unsigned long long ga = (unsigned long long)gptr;
  v4u g0 = { 1u,                                  // count=1, user mode, load
             ldsAddr,                             // lds_addr [63:32]
             (unsigned)ga,                        // global_addr[31:0]
             ((unsigned)(ga >> 32) & 0x01ffffffu) // global_addr[56:32]
               | 0x80000000u };                   // type=2 ("image")
  v8i g1 = { (int)(2u << 16),                     // data_size=2 (4 bytes)
             (int)((dim0 & 0xffffu) << 16),       // tensor_dim0[15:0] @ [63:48]
             (int)((dim0 >> 16) | ((dim1 & 0xffffu) << 16)),
             (int)((dim1 >> 16) | (tile0 << 16)), // tile_dim0 @ [127:112]
             (int)(tile1 & 0xffffu),              // tile_dim1; tile_dim2=0
             (int)stride0,                        // tensor_dim0_stride[31:0]
             0, 0 };
  v4i z4 = { 0, 0, 0, 0 };
#if defined(__clang_major__) && __clang_major__ >= 23
  v8i z8 = { 0, 0, 0, 0, 0, 0, 0, 0 };
  __builtin_amdgcn_tensor_load_to_lds(g0, g1, z4, z4, z8, 0);
#else
  __builtin_amdgcn_tensor_load_to_lds(g0, g1, z4, z4, 0);
#endif
}
#endif  // HAVE_TDM

// ---------------------------------------------------------------------------
// Generic fp32-in / fp32-out GEMM, bf16 WMMA core.  C[M,N] = A[M,K] * B[K,N].
// TRANSA: A stored [K,M] row-major (tile DMA'd in [k][m] layout).
// BIAS:   add bias[n].
// ROWSQ:  also emit rowsq[m] = sum_k A[m][k]^2  (requires gridDim.x == 1).
// Block = 256 threads (8 waves), tile 64x64, BK=32, each wave -> 16x32 of C.
// ---------------------------------------------------------------------------
template<bool TRANSA, bool BIAS, bool ROWSQ>
__global__ __launch_bounds__(256) void gemm_bf16_wmma(
    const float* __restrict__ A, const float* __restrict__ B,
    const float* __restrict__ bias, float* __restrict__ C,
    float* __restrict__ rowsq,
    int M, int N, int K, int lda, int ldb, int ldc)
{
  __shared__ float lA[2][64 * 32];   // normal: [m][k]; TRANSA: [k][m(64)]
  __shared__ float lB[2][32 * 64];   // [k][n]
  __shared__ float lsq[256];

  const int t    = threadIdx.x;
  const int wave = t >> 5, lane = t & 31;
  const int half = lane >> 4, l15 = lane & 15;
  const int m_blk = blockIdx.y * 64;
  const int n_blk = blockIdx.x * 64;
  const int m_off = (wave & 3) * 16;      // wave's M sub-tile
  const int n_off = (wave >> 2) * 32;     // wave's N sub-tile pair

  float sq = 0.f;
  v8f c0 = {}; v8f c1 = {};
  const int kIters = K >> 5;

#if HAVE_TDM
  auto issue = [&](int kt, int buf) {
    const int kb = kt << 5;
    if (TRANSA) {
      tdm_load_2d(lds_addr_of(&lA[buf][0]),
                  (const char*)A + ((size_t)kb * lda + m_blk) * 4u,
                  (unsigned)(M - m_blk), (unsigned)(K - kb),
                  (unsigned)lda, 64u, 32u);
    } else {
      tdm_load_2d(lds_addr_of(&lA[buf][0]),
                  (const char*)A + ((size_t)m_blk * lda + kb) * 4u,
                  (unsigned)(K - kb), (unsigned)(M - m_blk),
                  (unsigned)lda, 32u, 64u);
    }
    tdm_load_2d(lds_addr_of(&lB[buf][0]),
                (const char*)B + ((size_t)kb * ldb + n_blk) * 4u,
                (unsigned)(N - n_blk), (unsigned)(K - kb),
                (unsigned)ldb, 64u, 32u);
  };
  if (wave == 0) issue(0, 0);             // prologue: stage 0 in flight
#endif

  for (int kt = 0; kt < kIters; ++kt) {
    const int buf = kt & 1;
#if HAVE_TDM
    if (wave == 0) {                      // prefetch next, wait for current
      if (kt + 1 < kIters) {
        issue(kt + 1, buf ^ 1);
        __builtin_amdgcn_s_wait_tensorcnt(2);
      } else {
        __builtin_amdgcn_s_wait_tensorcnt(0);
      }
    }
#else
    {   // fallback: manual fp32 staging (same LDS layouts)
      const int kb = kt << 5;
      if (TRANSA) {
        const int kr = t >> 3, mc = (t & 7) * 8;
        for (int c = 0; c < 8; ++c) {
          const int mg = m_blk + mc + c;
          lA[buf][kr * 64 + mc + c] =
              (mg < M) ? A[(size_t)(kb + kr) * lda + mg] : 0.f;
        }
      } else {
        const int mr = t >> 2, kc = (t & 3) * 8;
        const int mg = m_blk + mr;
        for (int c = 0; c < 8; ++c)
          lA[buf][mr * 32 + kc + c] =
              (mg < M) ? A[(size_t)mg * lda + kb + kc + c] : 0.f;
      }
      const int kr = t >> 3, nc = (t & 7) * 8;
      for (int c = 0; c < 8; ++c) {
        const int ng = n_blk + nc + c;
        lB[buf][kr * 64 + nc + c] =
            (ng < N) ? B[(size_t)(kb + kr) * ldb + ng] : 0.f;
      }
    }
#endif
    __syncthreads();                      // tile kt visible to all waves

    if (ROWSQ) {                          // fixed row per thread, exact once
      const int mr = t >> 2, kc = (t & 3) * 8;
#pragma unroll
      for (int c = 0; c < 8; ++c) {
        const float v = lA[buf][mr * 32 + kc + c];
        sq += v * v;
      }
    }

    // fragments per wave32 16x16x32 bf16 VGPR layouts (fp32 LDS -> bf16 regs)
    union { v16bf v; unsigned u[8]; } af, b0f, b1f;
    const int mrow = m_off + l15;
#pragma unroll
    for (int d = 0; d < 8; ++d) {
      const int ka = (d < 4) ? (half * 8 + 2 * d) : (16 + half * 8 + 2 * (d - 4));
      float f0, f1;
      if (TRANSA) { f0 = lA[buf][ka * 64 + mrow]; f1 = lA[buf][(ka + 1) * 64 + mrow]; }
      else        { f0 = lA[buf][mrow * 32 + ka]; f1 = lA[buf][mrow * 32 + ka + 1]; }
      af.u[d] = (unsigned)f2bf(f0) | ((unsigned)f2bf(f1) << 16);
    }
    const int nrow0 = n_off + l15, nrow1 = n_off + 16 + l15;
#pragma unroll
    for (int d = 0; d < 8; ++d) {
      const int kk = (d < 4) ? (half * 8 + 2 * d) : (16 + half * 8 + 2 * (d - 4));
      b0f.u[d] = (unsigned)f2bf(lB[buf][kk * 64 + nrow0]) |
                 ((unsigned)f2bf(lB[buf][(kk + 1) * 64 + nrow0]) << 16);
      b1f.u[d] = (unsigned)f2bf(lB[buf][kk * 64 + nrow1]) |
                 ((unsigned)f2bf(lB[buf][(kk + 1) * 64 + nrow1]) << 16);
    }
    c0 = __builtin_amdgcn_wmma_f32_16x16x32_bf16(false, af.v, false, b0f.v,
                                                 (short)0, c0, false, false);
    c1 = __builtin_amdgcn_wmma_f32_16x16x32_bf16(false, af.v, false, b1f.v,
                                                 (short)0, c1, false, false);
    __syncthreads();                      // buf free for the next DMA
  }

  if (ROWSQ) {
    lsq[t] = sq;
    __syncthreads();
    if (t < 64) {
      const int mg = m_blk + t;
      if (mg < M)
        rowsq[mg] = lsq[4 * t] + lsq[4 * t + 1] + lsq[4 * t + 2] + lsq[4 * t + 3];
    }
  }

  // store C (M = r + half*8, N = lane&15 per 16x16 f32 D layout)
#pragma unroll
  for (int r = 0; r < 8; ++r) {
    const int mg  = m_blk + m_off + r + half * 8;
    const int ng0 = n_blk + n_off + l15;
    const int ng1 = n_blk + n_off + 16 + l15;
    if (mg < M) {
      if (ng0 < N) C[(size_t)mg * ldc + ng0] = c0[r] + (BIAS ? bias[ng0] : 0.f);
      if (ng1 < N) C[(size_t)mg * ldc + ng1] = c1[r] + (BIAS ? bias[ng1] : 0.f);
    }
  }
}

// ---------------------------------------------------------------------------
// G = concept^T @ concept (fp32, exact) + L_sparse_2 / norm_metrics scalars.
// ---------------------------------------------------------------------------
__global__ __launch_bounds__(256) void concept_gram(
    const float* __restrict__ cpt, float* __restrict__ G,
    float* __restrict__ out_s2, float* __restrict__ out_nm)
{
  __shared__ float ck[32 * 64];
  __shared__ float Gs[64 * 64];
  const int t  = threadIdx.x;
  const int i  = t >> 2;
  const int j0 = (t & 3) * 16;
  float acc[16];
#pragma unroll
  for (int j = 0; j < 16; ++j) acc[j] = 0.f;

  for (int kc = 0; kc < Dn; kc += 32) {
#pragma unroll
    for (int q = 0; q < 8; ++q) {
      const int idx = t * 8 + q;
      ck[idx] = cpt[(size_t)(kc + (idx >> 6)) * NCn + (idx & 63)];
    }
    __syncthreads();
    for (int kk = 0; kk < 32; ++kk) {
      const float a = ck[kk * 64 + i];
#pragma unroll
      for (int j = 0; j < 16; ++j) acc[j] += a * ck[kk * 64 + j0 + j];
    }
    __syncthreads();
  }
#pragma unroll
  for (int j = 0; j < 16; ++j) { Gs[i * 64 + j0 + j] = acc[j]; G[i * 64 + j0 + j] = acc[j]; }
  __syncthreads();
  if (t == 0) {   // fixed-order, deterministic
    float diag = 0.f, off = 0.f;
    for (int a = 0; a < 64; ++a)
      for (int b = 0; b < 64; ++b) {
        const float g = Gs[a * 64 + b];
        if (a == b) diag += g; else off += g;
      }
    *out_s2 = off / 4096.0f;
    *out_nm = diag / 4096.0f;
  }
}

// ---------------------------------------------------------------------------
// 64x64 Gauss-Jordan inverse (G is SPD -> no pivoting), fp32, one block.
// ---------------------------------------------------------------------------
__global__ __launch_bounds__(64) void invert64(
    const float* __restrict__ G, float* __restrict__ Ginv)
{
  __shared__ float a[64][128];
  const int t = threadIdx.x;
  for (int j = 0; j < 64; ++j) a[t][j] = G[t * 64 + j];
  for (int j = 0; j < 64; ++j) a[t][64 + j] = (j == t) ? 1.f : 0.f;
  __syncthreads();
  for (int p = 0; p < 64; ++p) {
    if (t == 0) {
      const float inv = 1.0f / a[p][p];
      for (int j = 0; j < 128; ++j) a[p][j] *= inv;
    }
    __syncthreads();
    if (t != p) {
      const float f = a[t][p];
      for (int j = 0; j < 128; ++j) a[t][j] -= f * a[p][j];
    }
    __syncthreads();
  }
  for (int j = 0; j < 64; ++j) Ginv[t * 64 + j] = a[t][64 + j];
}

// ---------------------------------------------------------------------------
// Small fp32 matmul (keeps the inverse chain in fp32): C[M,N] = A[M,K]B[K,N].
// ---------------------------------------------------------------------------
__global__ void matmul_f32_small(const float* __restrict__ A,
                                 const float* __restrict__ B,
                                 float* __restrict__ C, int M, int N, int K)
{
  const int idx = blockIdx.x * blockDim.x + threadIdx.x;
  if (idx >= M * N) return;
  const int m = idx / N, n = idx % N;
  float s = 0.f;
  for (int k = 0; k < K; ++k) s += A[m * K + k] * B[k * N + n];
  C[idx] = s;
}

// ---------------------------------------------------------------------------
// Per-concept top-50 smallest d2 + knn dot.  One block (128 thr) per concept.
// d2[n] = ||c||^2 - 2*R[n][c] + ||bank_n||^2 ; select on d2 (sqrt monotone).
// Deterministic: per-thread sorted lists + fixed-order k-way LDS merge.
// ---------------------------------------------------------------------------
__global__ __launch_bounds__(128) void topk_dot(
    const float* __restrict__ R, const float* __restrict__ bank_sq,
    const float* __restrict__ G, const float* __restrict__ bank,
    const float* __restrict__ cpt, float* __restrict__ dots, int N)
{
  __shared__ float sv[128 * KSEL];
  __shared__ int   si[128 * KSEL];
  __shared__ float rv[128];
  __shared__ int   rn[128];
  __shared__ int   rt[128];
  __shared__ int   sel[KSEL];
  __shared__ int   wtid;

  const int c = blockIdx.x;
  const int t = threadIdx.x;
  const float csq = G[c * 64 + c];          // con_sq = diag(G)

  float lv[KSEL]; int li[KSEL];
#pragma unroll
  for (int s = 0; s < KSEL; ++s) { lv[s] = FINF; li[s] = 0x7fffffff; }

  for (int n = t; n < N; n += 128) {
    const float d2 = csq - 2.0f * R[(size_t)n * NCn + c] + bank_sq[n];
    if (d2 < lv[KSEL - 1]) {
      int p = KSEL - 1;
      while (p > 0 && lv[p - 1] > d2) { lv[p] = lv[p - 1]; li[p] = li[p - 1]; --p; }
      lv[p] = d2; li[p] = n;
    }
  }
  for (int s = 0; s < KSEL; ++s) { sv[t * KSEL + s] = lv[s]; si[t * KSEL + s] = li[s]; }
  int myhead = 0;
  __syncthreads();

  for (int s = 0; s < KSEL; ++s) {
    rv[t] = (myhead < KSEL) ? sv[t * KSEL + myhead] : FINF;
    rn[t] = (myhead < KSEL) ? si[t * KSEL + myhead] : 0x7fffffff;
    rt[t] = t;
    __syncthreads();
    for (int off = 64; off > 0; off >>= 1) {
      if (t < off) {
        const float v2 = rv[t + off]; const int n2 = rn[t + off];
        if (v2 < rv[t] || (v2 == rv[t] && n2 < rn[t])) {
          rv[t] = v2; rn[t] = n2; rt[t] = rt[t + off];
        }
      }
      __syncthreads();
    }
    if (t == 0) { sel[s] = rn[0]; wtid = rt[0]; }
    __syncthreads();
    if (t == wtid) ++myhead;
    __syncthreads();
  }

  // knn dot: mean_k <concept_c, bank[sel[k]]>
  float acc = 0.f;
  for (int s = 0; s < KSEL; ++s) {
    const float* br = bank + (size_t)sel[s] * Dn;
    for (int d = t; d < Dn; d += 128) acc += cpt[(size_t)d * NCn + c] * br[d];
  }
  rv[t] = acc;
  __syncthreads();
  for (int off = 64; off > 0; off >>= 1) {
    if (t < off) rv[t] += rv[t + off];
    __syncthreads();
  }
  if (t == 0) dots[c] = rv[0] / (float)KSEL;
}

__global__ void finalize_s1(const float* __restrict__ dots, float* __restrict__ out)
{
  if (blockIdx.x == 0 && threadIdx.x == 0) {
    float s = 0.f;
    for (int c = 0; c < NCn; ++c) s += dots[c];
    *out = s / (float)NCn;
  }
}

// ---------------------------------------------------------------------------
extern "C" void kernel_launch(void* const* d_in, const int* in_sizes, int n_in,
                              void* d_out, int out_size, void* d_ws, size_t ws_size,
                              hipStream_t stream)
{
  const float* X    = (const float*)d_in[0];   // [8192,1024]
  const float* Cpt  = (const float*)d_in[1];   // [1024,64]
  const float* bank = (const float*)d_in[2];   // [100000,1024]
  const float* Wh   = (const float*)d_in[3];   // [1024,1000]
  const float* bh   = (const float*)d_in[4];   // [1000]
  // d_in[5] = topk (=50, hardcoded)

  float* out = (float*)d_out;
  float* orig_pred = out;                       // 8192*1000
  float* y_pred    = out + (size_t)BSn * Ccls;  // 8192*1000
  float* s1 = out + 2ull * BSn * Ccls;
  float* s2 = s1 + 1;
  float* nm = s2 + 1;

  // workspace layout (floats)
  float* ws   = (float*)d_ws;
  float* G    = ws;                 // 4096
  float* Ginv = G + 4096;           // 4096
  float* T1   = Ginv + 4096;        // 8192*64
  float* Wc   = T1 + (size_t)BSn * NCn;       // 64*1000
  float* M2   = Wc + (size_t)NCn * Ccls;      // 64*1000
  float* dots = M2 + (size_t)NCn * Ccls;      // 64
  float* bsq  = dots + NCn;                   // 100000
  float* R    = bsq + Nbank;                  // 100000*64

  // 1) G = C^T C (fp32) + scalar metrics
  concept_gram<<<1, 256, 0, stream>>>(Cpt, G, s2, nm);
  // 2) Ginv (fp32 Gauss-Jordan)
  invert64<<<1, 64, 0, stream>>>(G, Ginv);
  // 3) orig_pred = X @ W_h + b_h   (biggest GEMM, TDM + bf16 WMMA)
  gemm_bf16_wmma<false, true, false><<<dim3(16, 128), 256, 0, stream>>>(
      X, Wh, bh, orig_pred, nullptr, BSn, Ccls, Dn, Dn, Ccls, Ccls);
  // 4) T1 = X @ C
  gemm_bf16_wmma<false, false, false><<<dim3(1, 128), 256, 0, stream>>>(
      X, Cpt, nullptr, T1, nullptr, BSn, NCn, Dn, Dn, NCn, NCn);
  // 5) Wc = C^T @ W_h  (A transposed access into concept[1024,64])
  gemm_bf16_wmma<true, false, false><<<dim3(16, 1), 256, 0, stream>>>(
      Cpt, Wh, nullptr, Wc, nullptr, NCn, Ccls, Dn, NCn, Ccls, Ccls);
  // 6) M2 = Ginv @ Wc (small, fp32 to preserve inverse precision)
  matmul_f32_small<<<(NCn * Ccls + 255) / 256, 256, 0, stream>>>(
      Ginv, Wc, M2, NCn, Ccls, NCn);
  // 7) y_pred = T1 @ M2 + b_h
  gemm_bf16_wmma<false, true, false><<<dim3(16, 128), 256, 0, stream>>>(
      T1, M2, bh, y_pred, nullptr, BSn, Ccls, NCn, NCn, Ccls, Ccls);
  // 8) R = bank @ C, fused bank row sum-of-squares (bank read once)
  gemm_bf16_wmma<false, false, true><<<dim3(1, (Nbank + 63) / 64), 256, 0, stream>>>(
      bank, Cpt, nullptr, R, bsq, Nbank, NCn, Dn, Dn, NCn, NCn);
  // 9) per-concept top-50 + knn dot
  topk_dot<<<NCn, 128, 0, stream>>>(R, bsq, G, bank, Cpt, dots, Nbank);
  // 10) L_sparse_1
  finalize_s1<<<1, 32, 0, stream>>>(dots, s1);
}
